// MultiHeadAttention_17076789969468
// MI455X (gfx1250) — matrix operand order repaired and checked by reading, not compiled, not run on hardware
//
#include <hip/hip_runtime.h>

typedef __attribute__((ext_vector_type(16))) _Float16 v16h;
typedef __attribute__((ext_vector_type(8)))  _Float16 v8h;
typedef __attribute__((ext_vector_type(8)))  float    v8f;

#define S_LEN  2048
#define NH     16
#define DHEAD  64
#define DMODEL 1024
#define QKV_LD (3 * NH * DHEAD)   // 3072
#define MROWS  (2 * S_LEN)        // B*S = 4096

// ---------------------------------------------------------------------------
// One-time prep: f32 -> f16 elementwise (for x)
// ---------------------------------------------------------------------------
__global__ __launch_bounds__(256)
void cvt_f32_f16(const float* __restrict__ s, _Float16* __restrict__ d, int n)
{
    int i = blockIdx.x * 256 + threadIdx.x;
    if (i < n) d[i] = (_Float16)s[i];
}

// ---------------------------------------------------------------------------
// One-time prep: convert W[K,N] f32 -> f16 swizzled FRAGMENT-MAJOR:
//   Wf[((kp*(N/16) + nchunk)*32 + lane)*16 + klow]
// where k = kp*32 + (lane>>4)*16 + klow, n = nchunk*16 + (lane&15).
// A GEMM B panel (32k x 128n) becomes 8KB of contiguous memory; a wave's
// B fragment is one contiguous 32B LDS read.
// ---------------------------------------------------------------------------
__global__ __launch_bounds__(256)
void swizzle_w(const float* __restrict__ W, _Float16* __restrict__ Wf,
               int K, int N)
{
    int i = blockIdx.x * 256 + threadIdx.x;      // dest half index
    if (i >= K * N) return;
    int kl   = i & 15;
    int lane = (i >> 4) & 31;
    int nc   = (i >> 9) % (N >> 4);
    int kp   = i / ((N >> 4) << 9);
    int k = kp * 32 + (lane >> 4) * 16 + kl;
    int n = nc * 16 + (lane & 15);
    Wf[i] = (_Float16)W[(size_t)k * N + n];
}

// ---------------------------------------------------------------------------
// One-time prep: V slice of qkv -> Vt[b][h][d][s] (transposed, f16): the
// attention V^T A-fragment is two contiguous 16B groups per lane.
// ---------------------------------------------------------------------------
__global__ __launch_bounds__(256)
void transpose_v(const _Float16* __restrict__ qkv, _Float16* __restrict__ Vt)
{
    int i = blockIdx.x * 256 + threadIdx.x;      // dest index, B*H*DH*S total
    int s = i & (S_LEN - 1);
    int d = (i >> 11) & (DHEAD - 1);
    int h = (i >> 17) & (NH - 1);
    int b =  i >> 21;
    Vt[i] = qkv[(size_t)(b * S_LEN + s) * QKV_LD + 2 * NH * DHEAD + h * DHEAD + d];
}

// ---------------------------------------------------------------------------
// GEMM with fused bias via f16 WMMA:  C[M,N] = A[M,K] @ W[K,N] + bias[N]
// A: f16 row-major.  Wf: f16 fragment-major (see swizzle_w).
// 256 threads = 8 waves; tile 64x128; K stepped by 32.
// Panels staged with GLOBAL_LOAD_ASYNC_TO_LDS_B128, double-buffered so the
// next panel's DMA overlaps the current panel's WMMAs (ASYNCcnt ping-pong).
// ---------------------------------------------------------------------------
template <typename TOUT>
__global__ __launch_bounds__(256)
void gemm_f16wmma(const _Float16* __restrict__ A, const _Float16* __restrict__ Wf,
                  const float* __restrict__ bias, TOUT* __restrict__ C,
                  int M, int N, int K)
{
    __shared__ _Float16 As[2][64 * 32];        // ping-pong A panels (2x4KB)
    __shared__ _Float16 Bs[2][8 * 32 * 16];    // ping-pong B panels (2x8KB)

    const int t     = threadIdx.x;
    const int wave  = t >> 5;
    const int lane  = t & 31;
    const int llow  = lane & 15;
    const int lhalf = lane >> 4;

    const int m0  = blockIdx.y * 64;
    const int n0  = blockIdx.x * 128;
    const int mw  = (wave >> 1) * 16;
    const int nc0 = (wave & 1) * 4;

    const int arow = t >> 2;                   // A staging: 0..63
    const int acol = (t & 3) * 8;              // 0,8,16,24
    const int NC   = N >> 4;

    auto issue = [&](int k0, int buf) {
        unsigned long long ga =
            (unsigned long long)(A + (size_t)(m0 + arow) * K + k0 + acol);
        unsigned lda = (unsigned)(uintptr_t)&As[buf][arow * 32 + acol];
        asm volatile("global_load_async_to_lds_b128 %0, %1, off"
                     :: "v"(lda), "v"(ga) : "memory");
        const _Float16* gp =
            Wf + ((size_t)(k0 >> 5) * NC + (n0 >> 4)) * 512 + t * 16;
        unsigned ldb = (unsigned)(uintptr_t)&Bs[buf][t * 16];
        asm volatile("global_load_async_to_lds_b128 %0, %1, off"
                     :: "v"(ldb), "v"((unsigned long long)gp) : "memory");
        asm volatile("global_load_async_to_lds_b128 %0, %1, off"
                     :: "v"(ldb + 16u), "v"((unsigned long long)(gp + 8)) : "memory");
    };

    v8f acc[4] = {};
    const int NK = K >> 5;

    issue(0, 0);
    for (int it = 0; it < NK; ++it) {
        const int cur = it & 1;
        const int knext = (it + 1 < NK) ? (it + 1) * 32 : 0;  // last: dummy prefetch
        issue(knext, cur ^ 1);
        asm volatile("s_wait_asynccnt 0x3" ::: "memory");      // current panel ready
        __syncthreads();

        v16h af;
#pragma unroll
        for (int i = 0; i < 16; ++i) {
            int kk = lhalf * 16 + i;
            af[i] = As[cur][(mw + llow) * 32 + kk];
        }
#pragma unroll
        for (int j = 0; j < 4; ++j) {
            v16h bf = *(const v16h*)&Bs[cur][((nc0 + j) * 32 + lane) * 16];
            acc[j] = __builtin_amdgcn_wmma_f32_16x16x32_f16(
                false, af, false, bf, (short)0, acc[j], false, false);
        }
        __syncthreads();   // reads done before next iteration overwrites buf
    }

#pragma unroll
    for (int j = 0; j < 4; ++j) {
        int n = n0 + (nc0 + j) * 16 + llow;
        float bv = bias[n];
#pragma unroll
        for (int r = 0; r < 8; ++r) {
            int m = m0 + mw + r + lhalf * 8;
            C[(size_t)m * N + n] = (TOUT)(acc[j][r] + bv);
        }
    }
}

// ---------------------------------------------------------------------------
// Attention block step: 32 keys (FULL) or trailing 16 keys (!FULL).
// MASK=false for interior blocks (kb+31 <= q0): no causal compares.
// ctx accumulated TRANSPOSED (ctx^T = V^T @ P^T): each lane's accumulator
// column is its own query, so alpha-rescale and 1/l are lane-local scalars
// (no ds_bpermute). P^T B-fragment needs one half-wave shfl_xor exchange.
// ---------------------------------------------------------------------------
template <bool FULL, bool MASK>
__device__ __forceinline__ void attn_block(
    const _Float16* __restrict__ Kp, const _Float16* __restrict__ Vtp,
    int kb, int q0, int llow, int lhalf,
    const v16h (&qf)[2], v8f (&acc)[4], float& m_i, float& l_i)
{
    // ---- V^T A-fragments: chunk j = 16 dims x 32 keys; 2x16B per lane ----
    v16h va[4];
#pragma unroll
    for (int j = 0; j < 4; ++j) {
        const _Float16* vp = Vtp + (size_t)(j * 16 + llow) * S_LEN + kb + 16 * lhalf;
#pragma unroll
        for (int i = 0; i < 8; ++i)  va[j][i] = vp[i];
#pragma unroll
        for (int i = 8; i < 16; ++i) va[j][i] = vp[i];
    }

    // ---- scores: S^T = K @ Q^T ----
    const int NSB = FULL ? 2 : 1;
    v8f st[2] = {};
#pragma unroll
    for (int sb = 0; sb < NSB; ++sb) {
#pragma unroll
        for (int j = 0; j < 2; ++j) {
            v16h kf;
#pragma unroll
            for (int i = 0; i < 16; ++i) {
                int d = j * 32 + lhalf * 16 + i;
                kf[i] = Kp[(size_t)(kb + sb * 16 + llow) * QKV_LD + d];
            }
            st[sb] = __builtin_amdgcn_wmma_f32_16x16x32_f16(
                false, kf, false, qf[j], (short)0, st[sb], false, false);
        }
    }

    // ---- (optional) causal mask + online softmax ----
    const int NV = FULL ? 16 : 8;
    float sv[16];
    float mblk = -1e30f;
#pragma unroll
    for (int sb = 0; sb < NSB; ++sb)
#pragma unroll
        for (int r = 0; r < 8; ++r) {
            float s = st[sb][r];
            if (MASK) {
                int key = kb + sb * 16 + r + 8 * lhalf;
                s = (key <= q0 + llow) ? s : -1e9f;
            }
            sv[sb * 8 + r] = s;
            mblk = fmaxf(mblk, s);
        }
    mblk = fmaxf(mblk, __shfl_xor(mblk, 16, 32));
    float m_new = fmaxf(m_i, mblk);
    float alpha = __expf(m_i - m_new);
    float psum = 0.0f;
#pragma unroll
    for (int i = 0; i < NV; ++i) { sv[i] = __expf(sv[i] - m_new); psum += sv[i]; }
    psum += __shfl_xor(psum, 16, 32);
    l_i = l_i * alpha + psum;
    m_i = m_new;

    // ---- rescale ctx^T accumulators: lane-uniform alpha (query = llow) ----
#pragma unroll
    for (int j = 0; j < 4; ++j)
#pragma unroll
        for (int r = 0; r < 8; ++r) acc[j][r] *= alpha;

    // ---- P^T B-fragment: own half + half-wave exchange ----
    // lane owns keys {8*lhalf..+7} u {16+8*lhalf..+7}; B-layout needs
    // {16*lhalf..+15}. Send partner its missing 8, keep own 8.
    float recv[8];
#pragma unroll
    for (int i = 0; i < 8; ++i) {
        float send = (FULL && lhalf == 0) ? sv[8 + i] : sv[i];
        recv[i] = __shfl_xor(send, 16, 32);
    }
    v16h pb;
#pragma unroll
    for (int i = 0; i < 16; ++i) {
        float v;
        if (i < 8) v = lhalf ? recv[i] : sv[i];
        else       v = lhalf ? (FULL ? sv[i] : 0.0f) : recv[i - 8];
        if (!FULL && lhalf) v = 0.0f;
        pb[i] = (_Float16)v;
    }

    // ---- ctx^T += V^T @ P^T ----
#pragma unroll
    for (int j = 0; j < 4; ++j)
        acc[j] = __builtin_amdgcn_wmma_f32_16x16x32_f16(
            false, va[j], false, pb, (short)0, acc[j], false, false);
}

// ---------------------------------------------------------------------------
// Causal flash attention. One wave32 per 16-query tile.
// ---------------------------------------------------------------------------
__global__ __launch_bounds__(32)
void attn_kernel(const _Float16* __restrict__ qkv, const _Float16* __restrict__ Vt,
                 _Float16* __restrict__ ctx)
{
    const int lane  = threadIdx.x;
    const int llow  = lane & 15;
    const int lhalf = lane >> 4;

    const int tile = blockIdx.x;
    const int qb = tile % (S_LEN / 16);
    const int h  = (tile / (S_LEN / 16)) % NH;
    const int b  =  tile / ((S_LEN / 16) * NH);
    const int q0 = qb * 16;

    const size_t base = (size_t)b * S_LEN * QKV_LD + (size_t)h * DHEAD;
    const _Float16* Qp  = qkv + base + 0 * NH * DHEAD;
    const _Float16* Kp  = qkv + base + 1 * NH * DHEAD;
    const _Float16* Vtp = Vt + ((size_t)b * NH + h) * DHEAD * S_LEN;

    // Q^T B-fragments, pre-scaled by 1/sqrt(64)
    v16h qf[2];
#pragma unroll
    for (int j = 0; j < 2; ++j)
#pragma unroll
        for (int i = 0; i < 16; ++i) {
            int d = j * 32 + lhalf * 16 + i;
            qf[j][i] = Qp[(size_t)(q0 + llow) * QKV_LD + d] * (_Float16)0.125f;
        }

    v8f acc[4] = {};                 // ctx^T: rows=dims (j*16 + r + 8*lhalf),
    float m_i = -1e30f, l_i = 0.0f;  //        col = query llow

    const int kend  = q0 + 16;       // causal key limit (exclusive)
    const int nfree = q0 >> 5;       // mask-free full blocks (kb+31 <= q0)
    for (int it = 0; it < nfree; ++it)
        attn_block<true, false>(Kp, Vtp, it * 32, q0, llow, lhalf, qf, acc, m_i, l_i);
    // exactly one masked block remains: 32 keys if q0%32==16, else 16 keys
    if (kend - nfree * 32 == 32)
        attn_block<true, true >(Kp, Vtp, nfree * 32, q0, llow, lhalf, qf, acc, m_i, l_i);
    else
        attn_block<false, true>(Kp, Vtp, nfree * 32, q0, llow, lhalf, qf, acc, m_i, l_i);

    // ---- normalize (lane-local 1/l) and store: 8 consecutive dims/lane ----
    float linv = 1.0f / l_i;
    _Float16* Cp = ctx + ((size_t)b * S_LEN + q0 + llow) * (NH * DHEAD)
                       + (size_t)h * DHEAD + 8 * lhalf;
#pragma unroll
    for (int j = 0; j < 4; ++j) {
        v8h outv;
#pragma unroll
        for (int r = 0; r < 8; ++r) outv[r] = (_Float16)(acc[j][r] * linv);
        *reinterpret_cast<v8h*>(Cp + j * 16) = outv;
    }
}

// ---------------------------------------------------------------------------
extern "C" void kernel_launch(void* const* d_in, const int* in_sizes, int n_in,
                              void* d_out, int out_size, void* d_ws, size_t ws_size,
                              hipStream_t stream) {
    const float* x     = (const float*)d_in[0];
    const float* W_in  = (const float*)d_in[1];
    const float* b_in  = (const float*)d_in[2];
    const float* W_out = (const float*)d_in[3];
    const float* b_out = (const float*)d_in[4];
    // d_in[5] = causal mask (unused: applied analytically)
    float* out = (float*)d_out;

    // workspace layout (f16 halves)
    _Float16* qkv  = (_Float16*)d_ws;                        // 4096*3072
    _Float16* ctx  = qkv  + (size_t)MROWS * QKV_LD;          // 4096*1024
    _Float16* x_h  = ctx  + (size_t)MROWS * DMODEL;          // 4096*1024
    _Float16* Wi_f = x_h  + (size_t)MROWS * DMODEL;          // 1024*3072
    _Float16* Wo_f = Wi_f + (size_t)DMODEL * QKV_LD;         // 1024*1024
    _Float16* Vt   = Wo_f + (size_t)DMODEL * DMODEL;         // 2*16*64*2048

    // ---- one-time prep ----
    {
        int n = MROWS * DMODEL;
        cvt_f32_f16<<<dim3((n + 255) / 256), 256, 0, stream>>>(x, x_h, n);
    }
    {
        int n = DMODEL * QKV_LD;
        swizzle_w<<<dim3((n + 255) / 256), 256, 0, stream>>>(W_in, Wi_f, DMODEL, QKV_LD);
    }
    {
        int n = DMODEL * DMODEL;
        swizzle_w<<<dim3((n + 255) / 256), 256, 0, stream>>>(W_out, Wo_f, DMODEL, DMODEL);
    }

    // 1) QKV projection
    dim3 g1(QKV_LD / 128, MROWS / 64);
    gemm_f16wmma<_Float16>
        <<<g1, 256, 0, stream>>>(x_h, Wi_f, b_in, qkv, MROWS, QKV_LD, DMODEL);

    // 1b) V transpose for attention
    {
        int n = 2 * NH * DHEAD * S_LEN;
        transpose_v<<<dim3(n / 256), 256, 0, stream>>>(qkv, Vt);
    }

    // 2) causal flash attention
    attn_kernel<<<dim3(2 * NH * (S_LEN / 16)), 32, 0, stream>>>(qkv, Vt, ctx);

    // 3) output projection
    dim3 g2(DMODEL / 128, MROWS / 64);
    gemm_f16wmma<float>
        <<<g2, 256, 0, stream>>>(ctx, Wo_f, b_out, out, MROWS, DMODEL, DMODEL);
}
